// TorchMiniBlock_28123445854220
// MI455X (gfx1250) — compile-verified, matrix operand order checked by reading, not run on hardware
//
#include <hip/hip_runtime.h>
#include <math.h>

#define NB 8
#define NS 2048
#define ND 1024
#define QT 16
#define KTILE 32
#define NWAVE 8
#define DW 128

typedef __attribute__((ext_vector_type(16))) _Float16 v16h;
typedef __attribute__((ext_vector_type(4)))  _Float16 v4h;
typedef __attribute__((ext_vector_type(8)))  float    v8f;
typedef __attribute__((ext_vector_type(4)))  unsigned int u32x4;
typedef __attribute__((ext_vector_type(8)))  int i32x8;
typedef __attribute__((ext_vector_type(4)))  int i32x4;

struct __align__(16) u4 { unsigned int x, y, z, w; };
union frag16 { u4 q[2]; v16h h; };

__device__ __forceinline__ v16h load_frag(const void* row, int qsel) {
  const u4* p = (const u4*)row;
  frag16 f;
  f.q[0] = p[qsel];       // halves {0..7} or {8..15}
  f.q[1] = p[2 + qsel];   // halves {16..23} or {24..31}
  return f.h;
}

__device__ __forceinline__ int kbase(int j, int lane) {
  return ((j < 4) ? (2 * j) : (16 + 2 * (j - 4))) + ((lane & 16) ? 8 : 0);
}

__global__ __launch_bounds__(256, 1)
void cvt_f32_to_f16(const float* __restrict__ in, _Float16* __restrict__ out, int n4) {
  int i = blockIdx.x * blockDim.x + threadIdx.x;
  int stride = gridDim.x * blockDim.x;
  for (; i < n4; i += stride) {
    float4 v = ((const float4*)in)[i];
    v4h h;
    h[0] = (_Float16)v.x; h[1] = (_Float16)v.y;
    h[2] = (_Float16)v.z; h[3] = (_Float16)v.w;
    *(v4h*)(out + 4 * (size_t)i) = h;
  }
}

__global__ __launch_bounds__(256, 1)
void fused_attn_kernel(const _Float16* __restrict__ xh,
                       const _Float16* __restrict__ Wh,
                       float* __restrict__ out) {
  __shared__ _Float16     s_xt[KTILE][ND];        // 64 KB row-major K/V tile
  __shared__ unsigned int s_xtT[ND][KTILE / 2];   // 64 KB transposed, key-pairs packed
  __shared__ __align__(16) char s_dyn[32768];     // 32 KB: loop scratch / attn stash

  float (*s_part)[QT][KTILE] = (float (*)[QT][KTILE])(s_dyn);      // 16 KB
  float (*s_scores)[KTILE]   = (float (*)[KTILE])(s_dyn + 16384);  // 2 KB
  float* s_m    = (float*)(s_dyn + 18432);
  float* s_l    = (float*)(s_dyn + 18560);
  float* s_corr = (float*)(s_dyn + 18688);
  _Float16 (*s_attn)[ND] = (_Float16 (*)[ND])(s_dyn);              // post-loop only

  const int tid    = threadIdx.x;
  const int wave   = tid >> 5;
  const int lane   = tid & 31;
  const int ncol   = lane & 15;
  const int rowoff = (lane & 16) ? 8 : 0;
  const int qsel   = (lane & 16) ? 1 : 0;
  const int d0     = wave * DW;

  const int wg = blockIdx.x;
  const int b  = wg >> 7;
  const int q0 = (wg & 127) * QT;
  const _Float16* xb = xh + (size_t)b * NS * ND;
  const float scale = 0.03125f;
  const v8f vzero = {};

  v16h qf[4];
#pragma unroll
  for (int c = 0; c < 4; ++c)
    qf[c] = load_frag(&xb[(size_t)(q0 + ncol) * ND + d0 + 32 * c], qsel);

  if (tid < QT) { s_m[tid] = -INFINITY; s_l[tid] = 0.0f; }

  v8f O[8];
#pragma unroll
  for (int t = 0; t < 8; ++t) O[t] = vzero;

  const int nkt = (q0 + QT - 1) / KTILE + 1;
  for (int kt = 0; kt < nkt; ++kt) {
    const int k0 = kt * KTILE;
    __syncthreads();

#if defined(__HIP_DEVICE_COMPILE__) && __has_builtin(__builtin_amdgcn_tensor_load_to_lds)
    if (wave == 0) {
      unsigned long long ga = (unsigned long long)(const void*)(xb + (size_t)k0 * ND);
      unsigned int lds = (unsigned int)(unsigned long long)(void*)&s_xt[0][0];
      u32x4 g0;
      g0[0] = 1u;
      g0[1] = lds;
      g0[2] = (unsigned int)ga;
      g0[3] = (unsigned int)(ga >> 32) | 0x80000000u;   // type=2 (image)
      i32x8 g1;
      g1[0] = 1 << 16;              // data_size = 2 bytes
      g1[1] = (int)(1024u << 16);   // tensor_dim0 = 1024
      g1[2] = (int)(2048u << 16);   // tensor_dim1 = 2048
      g1[3] = (int)(1024u << 16);   // tile_dim0 = 1024
      g1[4] = 32;                   // tile_dim1 = 32
      g1[5] = 1024;                 // tensor_dim0_stride
      g1[6] = 0; g1[7] = 0;
      i32x4 z4 = {0, 0, 0, 0};
      i32x8 z8 = {0, 0, 0, 0, 0, 0, 0, 0};
      __builtin_amdgcn_tensor_load_to_lds(g0, g1, z4, z4, z8, 0);
      __builtin_amdgcn_s_wait_tensorcnt(0);
    }
#else
    for (int i = tid; i < KTILE * (ND / 8); i += 256) {
      int r = i >> 7, c = (i & 127) * 8;
      *(u4*)&s_xt[r][c] = *(const u4*)&xb[(size_t)(k0 + r) * ND + c];
    }
#endif
    __syncthreads();

    for (int i = tid; i < (KTILE / 2) * (ND / 8); i += 256) {
      int rp = i & 15, dch = i >> 4;
      u4 a  = *(const u4*)&s_xt[2 * rp][8 * dch];
      u4 b2 = *(const u4*)&s_xt[2 * rp + 1][8 * dch];
      unsigned int au[4] = {a.x, a.y, a.z, a.w};
      unsigned int bu[4] = {b2.x, b2.y, b2.z, b2.w};
#pragma unroll
      for (int w = 0; w < 4; ++w) {
        int d = 8 * dch + 2 * w;
        s_xtT[d][rp]     = (au[w] & 0xffffu) | (bu[w] << 16);
        s_xtT[d + 1][rp] = (au[w] >> 16) | (bu[w] & 0xffff0000u);
      }
    }
    __syncthreads();

    v8f c0 = vzero, c1 = vzero;
#pragma unroll
    for (int c = 0; c < 4; ++c) {
      v16h b0 = load_frag(&s_xt[ncol][d0 + 32 * c], qsel);
      v16h b1 = load_frag(&s_xt[16 + ncol][d0 + 32 * c], qsel);
      c0 = __builtin_amdgcn_wmma_f32_16x16x32_f16(false, qf[c], false, b0, (short)0, c0, false, false);
      c1 = __builtin_amdgcn_wmma_f32_16x16x32_f16(false, qf[c], false, b1, (short)0, c1, false, false);
    }
#pragma unroll
    for (int j = 0; j < 8; ++j) {
      s_part[wave][j + rowoff][ncol]      = c0[j];
      s_part[wave][j + rowoff][16 + ncol] = c1[j];
    }
    __syncthreads();

    for (int i = tid; i < QT * KTILE; i += 256) {
      int r = i / KTILE, kk = i % KTILE;
      float s = 0.0f;
#pragma unroll
      for (int w = 0; w < NWAVE; ++w) s += s_part[w][r][kk];
      s_scores[r][kk] = s * scale;
    }
    __syncthreads();

    if (tid < QT) {
      const int r = tid, q = q0 + r;
      float m_old = s_m[r], tmax = -INFINITY;
      for (int kk = 0; kk < KTILE; ++kk) {
        float sv = (k0 + kk <= q) ? s_scores[r][kk] : -INFINITY;
        tmax = fmaxf(tmax, sv);
      }
      float m_new = fmaxf(m_old, tmax);
      float corr  = __expf(m_old - m_new);
      float lsum  = 0.0f;
      for (int kk = 0; kk < KTILE; ++kk) {
        float p = (k0 + kk <= q) ? __expf(s_scores[r][kk] - m_new) : 0.0f;
        s_scores[r][kk] = p;
        lsum += p;
      }
      s_m[r] = m_new;
      s_l[r] = s_l[r] * corr + lsum;
      s_corr[r] = corr;
    }
    __syncthreads();

    float cr[8];
#pragma unroll
    for (int j = 0; j < 8; ++j) cr[j] = s_corr[j + rowoff];
    v16h pf;
#pragma unroll
    for (int j = 0; j < 8; ++j) {
      int k = kbase(j, lane);
      pf[2 * j]     = (_Float16)s_scores[ncol][k];
      pf[2 * j + 1] = (_Float16)s_scores[ncol][k + 1];
    }
#pragma unroll
    for (int t = 0; t < 8; ++t) {
#pragma unroll
      for (int j = 0; j < 8; ++j) O[t][j] *= cr[j];
      v16h bf = load_frag(&s_xtT[d0 + t * 16 + ncol][0], qsel);
      O[t] = __builtin_amdgcn_wmma_f32_16x16x32_f16(false, pf, false, bf, (short)0, O[t], false, false);
    }
  }

  float rinv[8];
#pragma unroll
  for (int j = 0; j < 8; ++j) rinv[j] = 1.0f / s_l[j + rowoff];
  __syncthreads();                     // s_dyn switches role to s_attn
#pragma unroll
  for (int t = 0; t < 8; ++t)
#pragma unroll
    for (int j = 0; j < 8; ++j)
      s_attn[j + rowoff][d0 + t * 16 + ncol] = (_Float16)(O[t][j] * rinv[j]);
  __syncthreads();

  v8f P[8];
#pragma unroll
  for (int t = 0; t < 8; ++t) P[t] = vzero;
  const int e0 = wave * DW;
  for (int kc = 0; kc < ND / 32; ++kc) {
    v16h af = load_frag(&s_attn[ncol][kc * 32], qsel);
#pragma unroll
    for (int t = 0; t < 8; ++t) {
      v16h bf = load_frag(&Wh[(size_t)(e0 + t * 16 + ncol) * ND + kc * 32], qsel);
      P[t] = __builtin_amdgcn_wmma_f32_16x16x32_f16(false, af, false, bf, (short)0, P[t], false, false);
    }
  }
  float* outb = out + ((size_t)b * NS + q0) * ND;
#pragma unroll
  for (int t = 0; t < 8; ++t)
#pragma unroll
    for (int j = 0; j < 8; ++j)
      outb[(size_t)(j + rowoff) * ND + e0 + t * 16 + ncol] = P[t][j];
}

extern "C" void kernel_launch(void* const* d_in, const int* in_sizes, int n_in,
                              void* d_out, int out_size, void* d_ws, size_t ws_size,
                              hipStream_t stream) {
  const float* x = (const float*)d_in[0];
  const float* W = (const float*)d_in[1];
  _Float16* xhf = (_Float16*)d_ws;                                     // 32 MB
  _Float16* Whf = (_Float16*)((char*)d_ws + (size_t)NB * NS * ND * 2); // +2 MB

  cvt_f32_to_f16<<<4096, 256, 0, stream>>>(x, xhf, NB * NS * ND / 4);
  cvt_f32_to_f16<<<1024, 256, 0, stream>>>(W, Whf, ND * ND / 4);
  fused_attn_kernel<<<NB * (NS / QT), 256, 0, stream>>>(xhf, Whf, (float*)d_out);
}
